// LlamaAttention_89584427860169
// MI455X (gfx1250) — matrix operand order repaired and checked by reading, not compiled, not run on hardware
//
#include <hip/hip_runtime.h>

// ---------------------------------------------------------------------------
// Llama attention layer for gfx1250 (MI455X).
// bf16 WMMA for all matmuls; all tile staging via Tensor Data Mover with
// double buffering; every LDS fragment read is ds_load_b128.
// Weights (and V) are pre-transposed once so no per-step transposes exist.
// B=1, S=2048, D=4096, H=32, KVH=8, HD=128
// ---------------------------------------------------------------------------

typedef unsigned short u16;
typedef unsigned int   u32;
typedef unsigned long long u64;

typedef __bf16 v16bf __attribute__((ext_vector_type(16)));
typedef float  v8f   __attribute__((ext_vector_type(8)));
typedef unsigned int uv4 __attribute__((ext_vector_type(4)));
typedef int          iv4 __attribute__((ext_vector_type(4)));
typedef int          iv8 __attribute__((ext_vector_type(8)));

#define SEQ    2048
#define DMODEL 4096
#define NH     32
#define NKVH   8
#define HDIM   128
#define HHD    4096   // NH*HDIM
#define KVHD   1024   // NKVH*HDIM

static __device__ __forceinline__ v8f zero8() {
  v8f z;
#pragma unroll
  for (int i = 0; i < 8; ++i) z[i] = 0.0f;
  return z;
}

// fp32 -> bf16 bits, round-to-nearest-even
static __device__ __forceinline__ u16 f2bf(float f) {
  u32 u = __builtin_bit_cast(u32, f);
  u32 r = u + 0x7FFFu + ((u >> 16) & 1u);
  return (u16)(r >> 16);
}

static __device__ __forceinline__ v8f wmma_bf16(v16bf a, v16bf b, v8f c) {
  return __builtin_amdgcn_wmma_f32_16x16x32_bf16(false, a, false, b,
                                                 (short)0, c, false, false);
}

// A-matrix fragment (16x32 bf16) from LDS row-major [m][k]:
// lane l: M = l&15, K-runs {kb..kb+7, kb+16..kb+23}, kb = (l>>4)*8.
static __device__ __forceinline__ v16bf load_a_frag(const u16* lds, int row0,
                                                    int k0, int ld) {
  int lane = threadIdx.x & 31;
  int row  = row0 + (lane & 15);
  int kb   = (lane >> 4) * 8;
  union { v16bf f; uint4 q[2]; } r;
  const u16* p = lds + row * ld + k0 + kb;
  r.q[0] = *(const uint4*)(p);
  r.q[1] = *(const uint4*)(p + 16);
  return r.f;
}

// B-matrix fragment (32x16 bf16) from LDS stored K-contiguous: lds[n][k].
// lane l: N = l&15, K-runs {kb..kb+7, kb+16..kb+23}, kb = (l>>4)*8 -> 2x ds_load_b128.
static __device__ __forceinline__ v16bf load_b_frag_nk(const u16* lds, int k0,
                                                       int n0, int ld) {
  int lane = threadIdx.x & 31;
  int col  = lane & 15;
  int kb   = (lane >> 4) * 8;
  union { v16bf f; uint4 q[2]; } r;
  const u16* p = lds + (n0 + col) * ld + k0 + kb;
  r.q[0] = *(const uint4*)(p);
  r.q[1] = *(const uint4*)(p + 16);
  return r.f;
}

// ---------------------------------------------------------------------------
// Tensor Data Mover: 2D tile (16-bit elements) global -> LDS.
// D# group0: count=1, lds_addr, global_addr(57b), type=2.
// D# group1: data_size=2B, tensor_dim0/1, tile_dim0/1, tensor_dim0_stride.
// ---------------------------------------------------------------------------
static __device__ __forceinline__ void tdm_load_2d_b16(
    u32 lds_off, const void* gptr, u32 tensor_d0, u32 tensor_d1,
    u64 stride0_elems, u32 tile_d0, u32 tile_d1) {
  u64 ga = (u64)(size_t)gptr;
  uv4 g0;
  g0[0] = 1u;                                     // count=1, no gather
  g0[1] = lds_off;                                // LDS byte address
  g0[2] = (u32)ga;                                // global addr [31:0]
  g0[3] = (u32)((ga >> 32) & 0x1FFFFFFu) | (2u << 30);  // [56:32] | type=2
  iv8 g1;
  g1[0] = (int)(1u << 16);                        // data_size = 1 (2 bytes)
  g1[1] = (int)((tensor_d0 & 0xFFFFu) << 16);     // tensor_dim0[15:0]
  g1[2] = (int)(((tensor_d0 >> 16) & 0xFFFFu) | ((tensor_d1 & 0xFFFFu) << 16));
  g1[3] = (int)(((tensor_d1 >> 16) & 0xFFFFu) | ((tile_d0 & 0xFFFFu) << 16));
  g1[4] = (int)(tile_d1 & 0xFFFFu);               // tile_dim1, tile_dim2=0
  g1[5] = (int)(u32)(stride0_elems & 0xFFFFFFFFull);
  g1[6] = (int)(u32)((stride0_elems >> 32) & 0xFFFFull);  // dim1_stride=0
  g1[7] = 0;
  iv4 g2 = {0, 0, 0, 0};
  iv4 g3 = {0, 0, 0, 0};
  iv8 g4 = {0, 0, 0, 0, 0, 0, 0, 0};
  __builtin_amdgcn_tensor_load_to_lds(g0, g1, g2, g3, g4, 0);
}

static __device__ __forceinline__ u32 lds_offset(const void* p) {
  return (u32)(size_t)p;  // low 32 bits of generic LDS address = LDS offset
}

// ---------------------------------------------------------------------------
// fp32 -> bf16 convert (plain)
// ---------------------------------------------------------------------------
__global__ void cvt_f32_bf16(const float* __restrict__ in,
                             u16* __restrict__ out, int n) {
  int i = blockIdx.x * blockDim.x + threadIdx.x;
  if (i < n) out[i] = f2bf(in[i]);
}

// ---------------------------------------------------------------------------
// Tiled transpose + fp32->bf16: in[R][C] -> out[C][R]. R,C multiples of 32.
// 32x32 tiles via LDS, coalesced on both sides. One-time pass per matrix.
// ---------------------------------------------------------------------------
__global__ __launch_bounds__(256) void cvt_transpose_bf16(
    const float* __restrict__ in, u16* __restrict__ out, int R, int C) {
  __shared__ float t[32][33];
  const int c0 = blockIdx.x * 32;
  const int r0 = blockIdx.y * 32;
  const int tx = threadIdx.x & 31;
  const int ty = threadIdx.x >> 5;  // 0..7
#pragma unroll
  for (int j = 0; j < 4; ++j) {
    int row = ty + j * 8;
    t[row][tx] = in[(size_t)(r0 + row) * C + c0 + tx];
  }
  __syncthreads();
#pragma unroll
  for (int j = 0; j < 4; ++j) {
    int row = ty + j * 8;
    out[(size_t)(c0 + row) * R + r0 + tx] = f2bf(t[tx][row]);
  }
}

// ---------------------------------------------------------------------------
// bf16 GEMM: C[M][N] (fp32) = A[M][K] * BT[N][K]^T  (BT = pre-transposed B).
// 128x128 block tile, 8 waves, each wave 32x64 (8 WMMAs / K-step).
// Both tiles staged by TDM, double-buffered; inner loop has no VMEM.
// M%128==0, N%128==0, K%32==0.
// ---------------------------------------------------------------------------
__global__ __launch_bounds__(256) void gemm_bf16_tn(const u16* __restrict__ A,
                                                    const u16* __restrict__ BT,
                                                    float* __restrict__ C,
                                                    int M, int N, int K) {
  const int m0   = blockIdx.y * 128;
  const int n0   = blockIdx.x * 128;
  const int tid  = threadIdx.x;
  const int wave = tid >> 5;
  const int lane = tid & 31;
  const int mw   = (wave & 3) * 32;
  const int nw   = (wave >> 2) * 64;

  __shared__ alignas(16) u16 sa[2][128 * 32];  // [m][k]
  __shared__ alignas(16) u16 sb[2][128 * 32];  // [n][k]

  v8f acc[2][4];
#pragma unroll
  for (int mi = 0; mi < 2; ++mi)
#pragma unroll
    for (int j = 0; j < 4; ++j) acc[mi][j] = zero8();

  u32 sa_off[2] = {lds_offset(sa[0]), lds_offset(sa[1])};
  u32 sb_off[2] = {lds_offset(sb[0]), lds_offset(sb[1])};

  auto issue = [&](int kt, int buf) {
    const int k0 = kt * 32;
    tdm_load_2d_b16(sa_off[buf], A + (size_t)m0 * K + k0, (u32)K, (u32)M,
                    (u64)K, 32, 128);
    tdm_load_2d_b16(sb_off[buf], BT + (size_t)n0 * K + k0, (u32)K, (u32)N,
                    (u64)K, 32, 128);
  };

  const int nk = K >> 5;
  if (wave == 0) {
    issue(0, 0);
    __builtin_amdgcn_s_wait_tensorcnt(0);
  }
  __syncthreads();

  for (int kt = 0; kt < nk; ++kt) {
    const int buf = kt & 1;
    if (wave == 0 && kt + 1 < nk) issue(kt + 1, buf ^ 1);  // overlap DMA

    v16bf a0 = load_a_frag(sa[buf], mw, 0, 32);
    v16bf a1 = load_a_frag(sa[buf], mw + 16, 0, 32);
#pragma unroll
    for (int j = 0; j < 4; ++j) {
      v16bf b = load_b_frag_nk(sb[buf], 0, nw + j * 16, 32);
      acc[0][j] = wmma_bf16(a0, b, acc[0][j]);
      acc[1][j] = wmma_bf16(a1, b, acc[1][j]);
    }

    if (wave == 0 && kt + 1 < nk) __builtin_amdgcn_s_wait_tensorcnt(0);
    __syncthreads();
  }

  const int col = lane & 15;
  const int hi  = lane >> 4;
#pragma unroll
  for (int mi = 0; mi < 2; ++mi)
#pragma unroll
    for (int r = 0; r < 8; ++r) {
      size_t row = (size_t)(m0 + mw + mi * 16 + 8 * hi + r) * N;
#pragma unroll
      for (int j = 0; j < 4; ++j)
        C[row + n0 + nw + j * 16 + col] = acc[mi][j][r];
    }
}

// ---------------------------------------------------------------------------
// RoPE: fp32 in [S][nCols] -> bf16 out. dims 0..63 rotated (pair d, d+32,
// freq index d), 64..127 passthrough.
// ---------------------------------------------------------------------------
__global__ void rope_bf16(const float* __restrict__ X,
                          const int* __restrict__ pos_ids,
                          u16* __restrict__ out, int nCols) {
  int i = blockIdx.x * blockDim.x + threadIdx.x;
  int total = SEQ * nCols;
  if (i >= total) return;
  int s = i / nCols;
  int d = i & (HDIM - 1);
  float x = X[i];
  float o;
  if (d < 64) {
    float pos = (float)pos_ids[s];
    float f = __expf(-(float)d * 0.14391156831212787f);  // ln(1e4)/64
    float a = pos * f;
    float ca = __cosf(a), sa = __sinf(a);
    if (d < 32) {
      o = x * ca - X[i + 32] * sa;
    } else {
      o = x * ca + X[i - 32] * sa;
    }
  } else {
    o = x;
  }
  out[i] = f2bf(o);
}

// ---------------------------------------------------------------------------
// Flash attention, causal. 64 queries/block, 4 waves, 32-key K/V tiles.
// Q, K, V(transposed) all staged via TDM; K/V double-buffered.
// Q: [S][HHD], K: [S][KVHD], VT: [KVHD][SEQ], O: [S][HHD] (all bf16).
// ---------------------------------------------------------------------------
__global__ __launch_bounds__(128) void flash_attn(const u16* __restrict__ Q,
                                                  const u16* __restrict__ Kb,
                                                  const u16* __restrict__ VT,
                                                  u16* __restrict__ O) {
  const int qbase = blockIdx.x * 64;
  const int h     = blockIdx.y;
  const int kvh   = h >> 2;  // H/KVH = 4
  const int tid   = threadIdx.x;
  const int wave  = tid >> 5;
  const int lane  = tid & 31;
  const int col   = lane & 15;
  const int hi    = lane >> 4;

  __shared__ alignas(16) u16 sq[64 * 128];       // [q][hd]
  __shared__ alignas(16) u16 sk[2][32 * 128];    // [key][hd] == [n][k]
  __shared__ alignas(16) u16 sv[2][128 * 32];    // [hd][key] == [n][k]
  __shared__ alignas(16) u16 sp[64 * 32];        // [q][key]  == [m][k]

  u32 sk_off[2] = {lds_offset(sk[0]), lds_offset(sk[1])};
  u32 sv_off[2] = {lds_offset(sv[0]), lds_offset(sv[1])};

  auto issueKV = [&](int kb, int buf) {
    const int kbase = kb * 32;
    tdm_load_2d_b16(sk_off[buf], Kb + (size_t)kbase * KVHD + kvh * HDIM,
                    KVHD, SEQ, KVHD, 128, 32);
    tdm_load_2d_b16(sv_off[buf], VT + (size_t)(kvh * HDIM) * SEQ + kbase,
                    SEQ, KVHD, SEQ, 32, 128);
  };

  const int nkb = (qbase >> 5) + 2;  // causal: skip blocks above diagonal
  if (wave == 0) {
    tdm_load_2d_b16(lds_offset(sq), Q + (size_t)qbase * HHD + h * HDIM,
                    HHD, SEQ, HHD, 128, 64);
    issueKV(0, 0);
    __builtin_amdgcn_s_wait_tensorcnt(0);
  }
  __syncthreads();

  // Q fragments: reused for every key block
  v16bf aq[4];
#pragma unroll
  for (int hk = 0; hk < 4; ++hk) aq[hk] = load_a_frag(sq, wave * 16, hk * 32, 128);

  float m[8], l[8];
  v8f acc[8];
#pragma unroll
  for (int r = 0; r < 8; ++r) { m[r] = -3.0e38f; l[r] = 0.0f; }
#pragma unroll
  for (int j = 0; j < 8; ++j) acc[j] = zero8();

  const float scale = 0.08838834764831845f;  // 1/sqrt(128)

  for (int kb = 0; kb < nkb; ++kb) {
    const int kbase = kb * 32;
    const int buf   = kb & 1;
    if (wave == 0 && kb + 1 < nkb) issueKV(kb + 1, buf ^ 1);  // overlap DMA

    // scores: 16 q-rows x 32 keys (accumulate over head dim, 4 x K=32)
    v8f c0 = zero8(), c1 = zero8();
#pragma unroll
    for (int hk = 0; hk < 4; ++hk) {
      v16bf b0 = load_b_frag_nk(sk[buf], hk * 32, 0, 128);   // keys 0..15
      v16bf b1 = load_b_frag_nk(sk[buf], hk * 32, 16, 128);  // keys 16..31
      c0 = wmma_bf16(aq[hk], b0, c0);
      c1 = wmma_bf16(aq[hk], b1, c1);
    }

    // online softmax (rows live in 16-lane halves)
    const int qrow_base = qbase + wave * 16 + 8 * hi;
#pragma unroll
    for (int r = 0; r < 8; ++r) {
      float s0 = c0[r] * scale;
      float s1 = c1[r] * scale;
      int qrow = qrow_base + r;
      if (kbase + col > qrow) s0 = -1.0e30f;
      if (kbase + 16 + col > qrow) s1 = -1.0e30f;

      float t = fmaxf(s0, s1);
      t = fmaxf(t, __shfl_xor(t, 1, 32));
      t = fmaxf(t, __shfl_xor(t, 2, 32));
      t = fmaxf(t, __shfl_xor(t, 4, 32));
      t = fmaxf(t, __shfl_xor(t, 8, 32));
      float mnew = fmaxf(m[r], t);
      float p0 = __expf(s0 - mnew);
      float p1 = __expf(s1 - mnew);
      float corr = __expf(m[r] - mnew);
      float ps = p0 + p1;
      ps += __shfl_xor(ps, 1, 32);
      ps += __shfl_xor(ps, 2, 32);
      ps += __shfl_xor(ps, 4, 32);
      ps += __shfl_xor(ps, 8, 32);
      l[r] = l[r] * corr + ps;
      m[r] = mnew;
#pragma unroll
      for (int j = 0; j < 8; ++j) acc[j][r] *= corr;

      int prow = (wave * 16 + 8 * hi + r) * 32;
      sp[prow + col]      = f2bf(p0);
      sp[prow + 16 + col] = f2bf(p1);
    }
    __syncthreads();  // sp visible before fragment gathers

    // O += P @ V
    v16bf ap = load_a_frag(sp, wave * 16, 0, 32);
#pragma unroll
    for (int j = 0; j < 8; ++j) {
      v16bf bv = load_b_frag_nk(sv[buf], 0, j * 16, 32);
      acc[j] = wmma_bf16(ap, bv, acc[j]);
    }

    if (wave == 0 && kb + 1 < nkb) __builtin_amdgcn_s_wait_tensorcnt(0);
    __syncthreads();  // next buffers ready; old buffers free for DMA
  }

  // normalize, store bf16
#pragma unroll
  for (int r = 0; r < 8; ++r) {
    float inv = 1.0f / l[r];
    size_t row = (size_t)(qbase + wave * 16 + 8 * hi + r) * HHD + h * HDIM;
#pragma unroll
    for (int j = 0; j < 8; ++j) O[row + j * 16 + col] = f2bf(acc[j][r] * inv);
  }
}

// ---------------------------------------------------------------------------
// host-side launch
// ---------------------------------------------------------------------------
extern "C" void kernel_launch(void* const* d_in, const int* in_sizes, int n_in,
                              void* d_out, int out_size, void* d_ws,
                              size_t ws_size, hipStream_t stream) {
  (void)in_sizes; (void)n_in; (void)out_size; (void)ws_size;
  const float* hidden = (const float*)d_in[0];
  // d_in[1] = dense causal mask -- applied analytically in flash_attn
  const int*   pos    = (const int*)d_in[2];
  const float* Wq     = (const float*)d_in[3];
  const float* Wk     = (const float*)d_in[4];
  const float* Wv     = (const float*)d_in[5];
  const float* Wo     = (const float*)d_in[6];
  float* out = (float*)d_out;

  char* ws = (char*)d_ws;
  size_t off = 0;
  auto alloc = [&](size_t bytes) -> char* {
    char* p = ws + off;
    off = (off + bytes + 255) & ~(size_t)255;
    return p;
  };

  u16*   Xb  = (u16*)alloc((size_t)SEQ * DMODEL * 2);
  u16*   WqT = (u16*)alloc((size_t)DMODEL * HHD * 2);   // [HHD][DMODEL]
  u16*   WkT = (u16*)alloc((size_t)DMODEL * KVHD * 2);  // [KVHD][DMODEL]
  u16*   WvT = (u16*)alloc((size_t)DMODEL * KVHD * 2);  // [KVHD][DMODEL]
  u16*   WoT = (u16*)alloc((size_t)HHD * DMODEL * 2);   // [DMODEL][HHD]
  float* Qf  = (float*)alloc((size_t)SEQ * HHD * 4);
  float* Kf  = (float*)alloc((size_t)SEQ * KVHD * 4);
  float* Vf  = (float*)alloc((size_t)SEQ * KVHD * 4);
  u16*   Qb  = (u16*)alloc((size_t)SEQ * HHD * 2);
  u16*   Kb2 = (u16*)alloc((size_t)SEQ * KVHD * 2);
  u16*   VT  = (u16*)alloc((size_t)SEQ * KVHD * 2);     // [KVHD][SEQ]
  u16*   Ob  = (u16*)alloc((size_t)SEQ * HHD * 2);

  // activations: plain convert; weights: transpose+convert (one-time)
  {
    int n = SEQ * DMODEL;
    cvt_f32_bf16<<<(n + 255) / 256, 256, 0, stream>>>(hidden, Xb, n);
  }
  cvt_transpose_bf16<<<dim3(HHD / 32, DMODEL / 32), 256, 0, stream>>>(
      Wq, WqT, DMODEL, HHD);
  cvt_transpose_bf16<<<dim3(KVHD / 32, DMODEL / 32), 256, 0, stream>>>(
      Wk, WkT, DMODEL, KVHD);
  cvt_transpose_bf16<<<dim3(KVHD / 32, DMODEL / 32), 256, 0, stream>>>(
      Wv, WvT, DMODEL, KVHD);
  cvt_transpose_bf16<<<dim3(DMODEL / 32, HHD / 32), 256, 0, stream>>>(
      Wo, WoT, HHD, DMODEL);

  // projections (A row-major, B pre-transposed)
  gemm_bf16_tn<<<dim3(HHD / 128, SEQ / 128), 256, 0, stream>>>(
      Xb, WqT, Qf, SEQ, HHD, DMODEL);
  gemm_bf16_tn<<<dim3(KVHD / 128, SEQ / 128), 256, 0, stream>>>(
      Xb, WkT, Kf, SEQ, KVHD, DMODEL);
  gemm_bf16_tn<<<dim3(KVHD / 128, SEQ / 128), 256, 0, stream>>>(
      Xb, WvT, Vf, SEQ, KVHD, DMODEL);

  // RoPE + bf16 pack; V transposed to [KVHD][SEQ] bf16
  {
    int nq = SEQ * HHD;
    rope_bf16<<<(nq + 255) / 256, 256, 0, stream>>>(Qf, pos, Qb, HHD);
    int nk = SEQ * KVHD;
    rope_bf16<<<(nk + 255) / 256, 256, 0, stream>>>(Kf, pos, Kb2, KVHD);
    cvt_transpose_bf16<<<dim3(KVHD / 32, SEQ / 32), 256, 0, stream>>>(
        Vf, VT, SEQ, KVHD);
  }

  // attention
  flash_attn<<<dim3(SEQ / 64, NH), 128, 0, stream>>>(Qb, Kb2, VT, Ob);

  // output projection -> fp32 d_out
  gemm_bf16_tn<<<dim3(DMODEL / 128, SEQ / 128), 256, 0, stream>>>(
      Ob, WoT, out, SEQ, DMODEL, HHD);
}
